// Sequence_39968965656589
// MI455X (gfx1250) — matrix operand order repaired and checked by reading, not compiled, not run on hardware
//
#include <hip/hip_runtime.h>

typedef _Float16 v16h __attribute__((ext_vector_type(16)));
typedef float    v8f  __attribute__((ext_vector_type(8)));
typedef unsigned int u32;

#define HDIM   51
#define TSEQ   512
#define BATCH  2048
// LDS layout (in dwords). Each weight matrix: 2 K-tiles x 16 N-tiles, each tile
// = 32 lanes x 8 dwords (16 f16 = one WMMA B fragment) = 256 dwords -> 8192 dwords.
#define OFF_WHH1 0
#define OFF_WIH2 8192
#define OFF_WHH2 16384
#define OFF_HB1  24576   // 16x64 f16 h1 staging (512 dwords)
#define OFF_HB2  25088   // 16x64 f16 h2 staging (512 dwords)
#define LDS_DWORDS 25600 // 100 KB dynamic LDS

__device__ __forceinline__ float sigm(float x) {
  return 1.0f / (1.0f + __expf(-x));
}
__device__ __forceinline__ float tanh_fast(float x) {
  // 1 - 2/(e^{2x}+1): monotone-safe at +/-inf (no inf/inf NaN)
  float e = __expf(2.0f * x);
  return 1.0f - 2.0f / (e + 1.0f);
}

// A-fragment (16x32 f16) K index for (lane-high-half, half-slot h):
// V0..V3 hold K 0..7 (lo lanes) / 8..15 (hi lanes), V4..V7 hold K 16..23 / 24..31.
__device__ __forceinline__ int kmapA(int hi, int h) {
  int v = h >> 1, p = h & 1;
  int kb = (v < 4) ? (2 * v) : (16 + 2 * (v - 4));
  return kb + (hi ? 8 : 0) + p;
}

__device__ __forceinline__ v16h loadBfrag(const u32* p) {
  union { uint4 q[2]; v16h h; } u;
  u.q[0] = ((const uint4*)p)[0];
  u.q[1] = ((const uint4*)p)[1];
  return u.h;
}

// Rebuild an A fragment (2 K-tiles of 16x32 f16) from an LDS 16x64 f16 row-major tile.
__device__ __forceinline__ void gatherA(const u32* hb32, int lane, v16h a[2]) {
  int m  = lane & 15;
  int hi = lane >> 4;
#pragma unroll
  for (int kt = 0; kt < 2; ++kt) {
#pragma unroll
    for (int hp = 0; hp < 8; ++hp) {
      int k = kt * 32 + kmapA(hi, 2 * hp);        // even -> dword aligned pair
      u32 u = hb32[(m * 64 + k) >> 1];
      union { u32 uu; _Float16 hh[2]; } pk; pk.uu = u;
      a[kt][2 * hp]     = pk.hh[0];
      a[kt][2 * hp + 1] = pk.hh[1];
    }
  }
}

__global__ void __launch_bounds__(128)
lstm2_wmma_kernel(const float* __restrict__ input,
                  const float* __restrict__ Wih1, const float* __restrict__ Whh1,
                  const float* __restrict__ bih1, const float* __restrict__ bhh1,
                  const float* __restrict__ Wih2, const float* __restrict__ Whh2,
                  const float* __restrict__ bih2, const float* __restrict__ bhh2,
                  const float* __restrict__ Wlin, const float* __restrict__ blin,
                  const int*   __restrict__ future,
                  float* __restrict__ out) {
  extern __shared__ u32 smem[];

  const int lane = threadIdx.x & 31;
  const int wv   = threadIdx.x >> 5;      // wave id 0..3, owns j = 16*wv .. 16*wv+15
  const int hi   = lane >> 4;             // high half of wave
  const int lm   = lane & 15;
  const int base = blockIdx.x * 16;       // batch rows base..base+15
  const int F    = future[0];
  const int TT   = TSEQ + F;

  // ---- One-time: pack W_hh1 / W_ih2 / W_hh2 into LDS in WMMA-B fragment order.
  // B tile (kt,nt): per lane 16 f16, lane-lo = K kt*32+{0..7,16..23}, lane-hi = K kt*32+{8..15,24..31},
  // N column = nt*16 + (lane&15). Gate-padded N space: gate = n>>6, j = n&63.
  for (int idx = wv; idx < 96; idx += 4) {
    int mi  = idx / 32;                   // 0:Whh1 1:Wih2 2:Whh2
    int rem = idx % 32;
    int kt  = rem >> 4, nt = rem & 15;
    const float* src = (mi == 0) ? Whh1 : (mi == 1) ? Wih2 : Whh2;
    u32* dst = smem + mi * 8192 + (kt * 16 + nt) * 256 + lane * 8;
    int n_pad = nt * 16 + lm;
    int gate  = n_pad >> 6;
    int j     = n_pad & 63;
    int row   = gate * HDIM + j;
#pragma unroll
    for (int h = 0; h < 16; h += 2) {
      int k = kt * 32 + (hi ? 8 : 0) + h + (h >= 8 ? 8 : 0);
      float f0 = (j < HDIM && (k    ) < HDIM) ? src[row * HDIM + k    ] : 0.f;
      float f1 = (j < HDIM && (k + 1) < HDIM) ? src[row * HDIM + k + 1] : 0.f;
      union { _Float16 hh[2]; u32 uu; } pk;
      pk.hh[0] = (_Float16)f0; pk.hh[1] = (_Float16)f1;
      dst[h >> 1] = pk.uu;
    }
  }

  // ---- Per-lane constants (this wave's column j across all 4 gates).
  const int jcol = wv * 16 + lm;
  float wih1c[4], bsum1[4], bsum2[4];
#pragma unroll
  for (int g = 0; g < 4; ++g) {
    int row = g * HDIM + jcol;
    bool v = (jcol < HDIM);
    wih1c[g] = v ? Wih1[row] : 0.f;
    bsum1[g] = v ? (bih1[row] + bhh1[row]) : 0.f;
    bsum2[g] = v ? (bih2[row] + bhh2[row]) : 0.f;
  }
  float wlA[2][16];
#pragma unroll
  for (int kt = 0; kt < 2; ++kt)
#pragma unroll
    for (int h = 0; h < 16; ++h) {
      int k = kt * 32 + kmapA(hi, h);
      wlA[kt][h] = (k < HDIM) ? Wlin[k] : 0.f;
    }
  const float blin0 = blin[0];

  u32*       hb1_32 = smem + OFF_HB1;
  u32*       hb2_32 = smem + OFF_HB2;
  _Float16*  hb1    = (_Float16*)hb1_32;
  _Float16*  hb2    = (_Float16*)hb2_32;

  // ---- Recurrent state in registers.
  float c1[8], c2[8];
  v16h a1[2], a2[2];
#pragma unroll
  for (int r = 0; r < 8; ++r) { c1[r] = 0.f; c2[r] = 0.f; }
#pragma unroll
  for (int kt = 0; kt < 2; ++kt)
#pragma unroll
    for (int h = 0; h < 16; ++h) { a1[kt][h] = (_Float16)0.f; a2[kt][h] = (_Float16)0.f; }

  float xnext[8];
#pragma unroll
  for (int r = 0; r < 8; ++r) xnext[r] = 0.f;

  __syncthreads();   // weights ready

  for (int t = 0; t < TT; ++t) {
    // x_t per C/D row m = r + 8*hi
    float xv[8];
    if (t < TSEQ) {
#pragma unroll
      for (int r = 0; r < 8; ++r)
        xv[r] = input[(size_t)(base + r + 8 * hi) * TSEQ + t];
    } else {
#pragma unroll
      for (int r = 0; r < 8; ++r) xv[r] = xnext[r];
    }

    // ---- Layer 1: z1 = bias + x*W_ih1 + h1 @ W_hh1^T
    v8f acc[4];
#pragma unroll
    for (int g = 0; g < 4; ++g)
#pragma unroll
      for (int r = 0; r < 8; ++r)
        acc[g][r] = bsum1[g] + xv[r] * wih1c[g];
#pragma unroll
    for (int kt = 0; kt < 2; ++kt)
#pragma unroll
      for (int g = 0; g < 4; ++g) {
        v16h bf = loadBfrag(smem + OFF_WHH1 + ((kt * 16 + (g * 4 + wv)) * 256 + lane * 8));
        acc[g] = __builtin_amdgcn_wmma_f32_16x16x32_f16(
            false, a1[kt], false, bf, (short)0, acc[g], false, false);
      }
    // activations -> c1, new h1 (lane-local thanks to gate-padded N layout)
#pragma unroll
    for (int r = 0; r < 8; ++r) {
      float iv = sigm(acc[0][r]);
      float fv = sigm(acc[1][r]);
      float gv = tanh_fast(acc[2][r]);
      float ov = sigm(acc[3][r]);
      float cn = fv * c1[r] + iv * gv;
      c1[r] = cn;
      float hv = ov * tanh_fast(cn);
      hb1[(r + 8 * hi) * 64 + jcol] = (_Float16)hv;
    }
    __syncthreads();
    gatherA(hb1_32, lane, a1);

    // ---- Layer 2: z2 = bias + h1 @ W_ih2^T + h2 @ W_hh2^T
#pragma unroll
    for (int g = 0; g < 4; ++g)
#pragma unroll
      for (int r = 0; r < 8; ++r)
        acc[g][r] = bsum2[g];
#pragma unroll
    for (int kt = 0; kt < 2; ++kt)
#pragma unroll
      for (int g = 0; g < 4; ++g) {
        v16h bf = loadBfrag(smem + OFF_WIH2 + ((kt * 16 + (g * 4 + wv)) * 256 + lane * 8));
        acc[g] = __builtin_amdgcn_wmma_f32_16x16x32_f16(
            false, a1[kt], false, bf, (short)0, acc[g], false, false);
      }
#pragma unroll
    for (int kt = 0; kt < 2; ++kt)
#pragma unroll
      for (int g = 0; g < 4; ++g) {
        v16h bf = loadBfrag(smem + OFF_WHH2 + ((kt * 16 + (g * 4 + wv)) * 256 + lane * 8));
        acc[g] = __builtin_amdgcn_wmma_f32_16x16x32_f16(
            false, a2[kt], false, bf, (short)0, acc[g], false, false);
      }
#pragma unroll
    for (int r = 0; r < 8; ++r) {
      float iv = sigm(acc[0][r]);
      float fv = sigm(acc[1][r]);
      float gv = tanh_fast(acc[2][r]);
      float ov = sigm(acc[3][r]);
      float cn = fv * c2[r] + iv * gv;
      c2[r] = cn;
      float hv = ov * tanh_fast(cn);
      hb2[(r + 8 * hi) * 64 + jcol] = (_Float16)hv;
    }
    __syncthreads();
    gatherA(hb2_32, lane, a2);

    // ---- Projection out = h2 @ W_lin^T + b_lin, from A-layout fragment.
    float s = 0.f;
#pragma unroll
    for (int kt = 0; kt < 2; ++kt)
#pragma unroll
      for (int h = 0; h < 16; ++h)
        s += (float)a2[kt][h] * wlA[kt][h];
    s += __shfl_xor(s, 16, 32);        // lane pair (l, l+16) covers all K of row m=l&15
    float outv = s + blin0;
    if (wv == 0 && lane < 16)
      out[(size_t)(base + lane) * TT + t] = outv;
    // closed-loop feedback: every lane grabs x for its C/D rows
#pragma unroll
    for (int r = 0; r < 8; ++r)
      xnext[r] = __shfl(outv, r + 8 * hi, 32);
  }
}

extern "C" void kernel_launch(void* const* d_in, const int* in_sizes, int n_in,
                              void* d_out, int out_size, void* d_ws, size_t ws_size,
                              hipStream_t stream) {
  const float* input = (const float*)d_in[0];
  const float* Wih1  = (const float*)d_in[1];
  const float* Whh1  = (const float*)d_in[2];
  const float* bih1  = (const float*)d_in[3];
  const float* bhh1  = (const float*)d_in[4];
  const float* Wih2  = (const float*)d_in[5];
  const float* Whh2  = (const float*)d_in[6];
  const float* bih2  = (const float*)d_in[7];
  const float* bhh2  = (const float*)d_in[8];
  const float* Wlin  = (const float*)d_in[9];
  const float* blin  = (const float*)d_in[10];
  const int*   fut   = (const int*)d_in[11];
  float* out = (float*)d_out;

  dim3 grid(BATCH / 16);
  dim3 block(128);
  size_t shmem = (size_t)LDS_DWORDS * 4;   // 100 KB dynamic LDS
  lstm2_wmma_kernel<<<grid, block, shmem, stream>>>(
      input, Wih1, Whh1, bih1, bhh1, Wih2, Whh2, bih2, bhh2, Wlin, blin, fut, out);
}
